// GraphTransformer_15960098472480
// MI455X (gfx1250) — matrix operand order repaired and checked by reading, not compiled, not run on hardware
//
#include <hip/hip_runtime.h>
#include <math.h>

typedef __attribute__((ext_vector_type(16))) _Float16 v16h;
typedef __attribute__((ext_vector_type(8)))  float    v8f;
typedef __attribute__((ext_vector_type(4)))  _Float16 v4h;

#define GDIM 128
#define LNEPS 1e-5f
#define APAD 34   // padded LDS row stride (halfs) for A tile [64][32]
#define BPAD 34   // padded LDS row stride (halfs) for B^T tile [128][32]

__device__ __forceinline__ float wave_sum32(float v) {
#pragma unroll
  for (int off = 16; off > 0; off >>= 1) v += __shfl_xor(v, off, 32);
  return v;
}

// ---------------- f32 -> f16 convert (weights) ----------------
__global__ void k_f32_to_f16(const float* __restrict__ in, _Float16* __restrict__ out, int n) {
  int i = blockIdx.x * blockDim.x + threadIdx.x;
  if (i < n) out[i] = (_Float16)in[i];
}

// ---------------- init mx/z/num ----------------
__global__ void k_init(float* __restrict__ mx, float* __restrict__ z,
                       float* __restrict__ num, int n) {
  int i = blockIdx.x * blockDim.x + threadIdx.x;
  if (i < n) { mx[i] = -3.0e38f; z[i] = 0.0f; num[i] = 0.0f; }
}

// ---------------- fused double LayerNorm: s = LN(LN(h,w0,b0),w1,b1) -> f16 ----------------
__global__ void k_ln2(const float* __restrict__ h,
                      const float* __restrict__ w0, const float* __restrict__ b0,
                      const float* __restrict__ w1, const float* __restrict__ b1,
                      _Float16* __restrict__ sout, int N) {
  int wid = threadIdx.x >> 5, lane = threadIdx.x & 31;
  int row = blockIdx.x * 8 + wid;
  if (row >= N) return;
  const float* hr = h + (size_t)row * GDIM;
  float x[4];
#pragma unroll
  for (int i = 0; i < 4; ++i) x[i] = hr[lane * 4 + i];
  float sum = x[0] + x[1] + x[2] + x[3];
  float sq  = x[0]*x[0] + x[1]*x[1] + x[2]*x[2] + x[3]*x[3];
  sum = wave_sum32(sum); sq = wave_sum32(sq);
  float mean = sum * (1.0f / GDIM);
  float var  = sq * (1.0f / GDIM) - mean * mean;
  float rs   = rsqrtf(var + LNEPS);
#pragma unroll
  for (int i = 0; i < 4; ++i) {
    int c = lane * 4 + i;
    x[i] = (x[i] - mean) * rs * w0[c] + b0[c];
  }
  sum = x[0] + x[1] + x[2] + x[3];
  sq  = x[0]*x[0] + x[1]*x[1] + x[2]*x[2] + x[3]*x[3];
  sum = wave_sum32(sum); sq = wave_sum32(sq);
  mean = sum * (1.0f / GDIM);
  var  = sq * (1.0f / GDIM) - mean * mean;
  rs   = rsqrtf(var + LNEPS);
  v4h sv;
#pragma unroll
  for (int i = 0; i < 4; ++i) {
    int c = lane * 4 + i;
    sv[i] = (_Float16)((x[i] - mean) * rs * w1[c] + b1[c]);
  }
  *(v4h*)(sout + (size_t)row * GDIM + lane * 4) = sv;
}

// ---------------- finalize: rst = num/z, y = LN(rst) -> f16 ----------------
__global__ void k_finalize(const float* __restrict__ num, const float* __restrict__ z,
                           const float* __restrict__ w, const float* __restrict__ b,
                           _Float16* __restrict__ yout, int N) {
  int wid = threadIdx.x >> 5, lane = threadIdx.x & 31;
  int row = blockIdx.x * 8 + wid;
  if (row >= N) return;
  size_t base = (size_t)row * GDIM + lane * 4;
  float x[4];
#pragma unroll
  for (int i = 0; i < 4; ++i) {
    float zz = z[base + i];
    x[i] = (zz > 0.0f) ? num[base + i] / zz : 0.0f;
  }
  float sum = x[0] + x[1] + x[2] + x[3];
  float sq  = x[0]*x[0] + x[1]*x[1] + x[2]*x[2] + x[3]*x[3];
  sum = wave_sum32(sum); sq = wave_sum32(sq);
  float mean = sum * (1.0f / GDIM);
  float var  = sq * (1.0f / GDIM) - mean * mean;
  float rs   = rsqrtf(var + LNEPS);
  v4h sv;
#pragma unroll
  for (int i = 0; i < 4; ++i) {
    int c = lane * 4 + i;
    sv[i] = (_Float16)((x[i] - mean) * rs * w[c] + b[c]);
  }
  *(v4h*)(yout + (size_t)row * GDIM + lane * 4) = sv;
}

// ---------------- edge pass 1: per-channel segment max via float atomicMax ----------------
__global__ void k_edge_max(const float* __restrict__ q, const float* __restrict__ kk,
                           const int* __restrict__ src, const int* __restrict__ dst,
                           float* __restrict__ mx, int E) {
  int wid = threadIdx.x >> 5, lane = threadIdx.x & 31;
  int e = blockIdx.x * 8 + wid;
  if (e >= E) return;
  int s = src[e], d = dst[e];
  const float4* qr = (const float4*)(q  + (size_t)s * GDIM);
  const float4* kr = (const float4*)(kk + (size_t)d * GDIM);
  float4 qv = qr[lane], kv = kr[lane];
  float a0 = qv.x * kv.x, a1 = qv.y * kv.y, a2 = qv.z * kv.z, a3 = qv.w * kv.w;
  float* m = mx + (size_t)d * GDIM + lane * 4;
  atomicMax(m + 0, a0); atomicMax(m + 1, a1);
  atomicMax(m + 2, a2); atomicMax(m + 3, a3);
}

// ---------------- edge pass 2: e = exp(att - mx[dst]); z += e; num += v[src]*e ----------------
__global__ void k_edge_acc(const float* __restrict__ q, const float* __restrict__ kk,
                           const float* __restrict__ v,
                           const int* __restrict__ src, const int* __restrict__ dst,
                           const float* __restrict__ mx,
                           float* __restrict__ z, float* __restrict__ num, int E) {
  int wid = threadIdx.x >> 5, lane = threadIdx.x & 31;
  int e = blockIdx.x * 8 + wid;
  if (e >= E) return;
  int s = src[e], d = dst[e];
  const float4* qr = (const float4*)(q  + (size_t)s * GDIM);
  const float4* kr = (const float4*)(kk + (size_t)d * GDIM);
  const float4* vr = (const float4*)(v  + (size_t)s * GDIM);
  const float4* mr = (const float4*)(mx + (size_t)d * GDIM);
  float4 qv = qr[lane], kv = kr[lane], vv = vr[lane], mv = mr[lane];
  float e0 = __expf(qv.x * kv.x - mv.x);
  float e1 = __expf(qv.y * kv.y - mv.y);
  float e2 = __expf(qv.z * kv.z - mv.z);
  float e3 = __expf(qv.w * kv.w - mv.w);
  size_t base = (size_t)d * GDIM + lane * 4;
  atomicAdd(z + base + 0, e0); atomicAdd(z + base + 1, e1);
  atomicAdd(z + base + 2, e2); atomicAdd(z + base + 3, e3);
  atomicAdd(num + base + 0, vv.x * e0); atomicAdd(num + base + 1, vv.y * e1);
  atomicAdd(num + base + 2, vv.z * e2); atomicAdd(num + base + 3, vv.w * e3);
}

// ---------------- WMMA GEMM: C[N,Nc] = A[N,K] @ W[K,Nc], fused epilogues ----------------
// Block tile 64M x 128N, K-step 32. 8 waves as 2(M) x 4(N); wave tile 32x32 = 2x2 WMMA accs.
// mode 0: split to q (*1/sqrt(128)), k, v  (o0,o1,o2)
// mode 1: bias + exact gelu -> f16 (o16)
// mode 2: bias + residual -> f32 (o0)
__global__ __launch_bounds__(256)
void k_wmma_gemm(const _Float16* __restrict__ A, const _Float16* __restrict__ W,
                 int N, int K, int Nc, int mode,
                 const float* __restrict__ bias, const float* __restrict__ resid,
                 float* __restrict__ o0, float* __restrict__ o1, float* __restrict__ o2,
                 _Float16* __restrict__ o16) {
  __shared__ __align__(16) _Float16 As[64 * APAD];    // [64 rows][32 k], padded
  __shared__ __align__(16) _Float16 Bs[128 * BPAD];   // transposed: [128 n][32 k], padded
  int tid = threadIdx.x;
  int lane = tid & 31, wid = tid >> 5;
  int wave_m = wid & 1, wave_n = wid >> 1;            // 2 x 4 waves
  int blockM = blockIdx.x * 64;
  int blockN = blockIdx.y * 128;
  v8f acc[2][2] = {{{}, {}}, {{}, {}}};

  const int m_l   = lane & 15;
  const int hi    = (lane >> 4) & 1;
  const int khalf = hi * 8;     // unused; A frags also use kbase
  const int kbase = hi * 16;    // B-frag K offset pattern

  for (int k0 = 0; k0 < K; k0 += 32) {
    // ---- stage A tile: 2048 halfs, 8 per thread, coalesced 16B reads ----
    {
      int idx = tid * 8;
      int r = idx >> 5, c = idx & 31;
      int grow = blockM + r;
      uint4 vv = make_uint4(0, 0, 0, 0);
      if (grow < N) vv = *(const uint4*)(A + (size_t)grow * K + k0 + c);
      uint* dst4 = (uint*)&As[r * APAD + c];
      dst4[0] = vv.x; dst4[1] = vv.y; dst4[2] = vv.z; dst4[3] = vv.w;
    }
    // ---- stage B tile transposed: 32k x 128n = 4096 halfs, 16 per thread ----
    {
      int idx = tid * 16;
      int k = idx >> 7, n = idx & 127;                 // n multiple of 16
      const uint4* wp = (const uint4*)(W + (size_t)(k0 + k) * Nc + blockN + n);
      uint4 w0 = wp[0], w1 = wp[1];                    // 16 halfs along n, coalesced
      const _Float16* h0 = (const _Float16*)&w0;
      const _Float16* h1 = (const _Float16*)&w1;
#pragma unroll
      for (int i = 0; i < 8; ++i) Bs[(n + i) * BPAD + k] = h0[i];
#pragma unroll
      for (int i = 0; i < 8; ++i) Bs[(n + 8 + i) * BPAD + k] = h1[i];
    }
    __syncthreads();

    // ---- fragments ----
    union { v16h v; uint u[8]; } af[2], bf[2];
#pragma unroll
    for (int i = 0; i < 2; ++i) {                      // A frags: rows wave_m*32 + i*16 + m
      const _Float16* Ar = &As[(wave_m * 32 + i * 16 + m_l) * APAD];
#pragma unroll
      for (int j = 0; j < 8; ++j) {
        int kb = kbase + 2 * j;
        af[i].u[j] = *(const uint*)(Ar + kb);
      }
    }
#pragma unroll
    for (int j = 0; j < 2; ++j) {                      // B frags: cols wave_n*32 + j*16 + n
      const _Float16* Br = &Bs[(wave_n * 32 + j * 16 + m_l) * BPAD + kbase];
#pragma unroll
      for (int t = 0; t < 8; ++t) bf[j].u[t] = *(const uint*)(Br + 2 * t);
    }
    // ---- 4 back-to-back WMMAs, reusing both fragment pairs ----
#pragma unroll
    for (int i = 0; i < 2; ++i)
#pragma unroll
      for (int j = 0; j < 2; ++j)
        acc[i][j] = __builtin_amdgcn_wmma_f32_16x16x32_f16(
            false, af[i].v, false, bf[j].v, (short)0, acc[i][j], false, false);
    __syncthreads();
  }

  // ---- epilogue ----
#pragma unroll
  for (int i = 0; i < 2; ++i) {
    int m_base = blockM + wave_m * 32 + i * 16 + hi * 8;
#pragma unroll
    for (int j = 0; j < 2; ++j) {
      int n_g = blockN + wave_n * 32 + j * 16 + m_l;
#pragma unroll
      for (int r = 0; r < 8; ++r) {
        int m_g = m_base + r;
        if (m_g >= N) continue;
        float val = acc[i][j][r];
        if (mode == 0) {
          if (n_g < 128)      o0[(size_t)m_g * 128 + n_g]         = val * 0.08838834764831845f;
          else if (n_g < 256) o1[(size_t)m_g * 128 + (n_g - 128)] = val;
          else                o2[(size_t)m_g * 128 + (n_g - 256)] = val;
        } else if (mode == 1) {
          float x = val + bias[n_g];
          float g = 0.5f * x * (1.0f + erff(x * 0.7071067811865475f));
          o16[(size_t)m_g * Nc + n_g] = (_Float16)g;
        } else {
          o0[(size_t)m_g * 128 + n_g] = val + bias[n_g] + resid[(size_t)m_g * 128 + n_g];
        }
      }
    }
  }
}

extern "C" void kernel_launch(void* const* d_in, const int* in_sizes, int n_in,
                              void* d_out, int out_size, void* d_ws, size_t ws_size,
                              hipStream_t stream) {
  const float* h0     = (const float*)d_in[0];
  const int*   src    = (const int*)d_in[1];
  const int*   dst    = (const int*)d_in[2];
  const float* norm_w = (const float*)d_in[3];
  const float* norm_b = (const float*)d_in[4];
  const float* nin_w  = (const float*)d_in[5];
  const float* nin_b  = (const float*)d_in[6];
  const float* Wqkv   = (const float*)d_in[7];
  const float* no_w   = (const float*)d_in[8];
  const float* no_b   = (const float*)d_in[9];
  const float* W1     = (const float*)d_in[10];
  const float* b1     = (const float*)d_in[11];
  const float* W2     = (const float*)d_in[12];
  const float* b2     = (const float*)d_in[13];

  const int N = in_sizes[0] / GDIM;
  const int E = in_sizes[1];

  char* p = (char*)d_ws;
  auto take = [&](size_t bytes) -> char* {
    char* r = p; p += (bytes + 255) & ~(size_t)255; return r;
  };
  _Float16* wqkv16 = (_Float16*)take((size_t)128 * 384 * 2);
  _Float16* w116   = (_Float16*)take((size_t)128 * 512 * 2);
  _Float16* w216   = (_Float16*)take((size_t)512 * 128 * 2);
  _Float16* s16    = (_Float16*)take((size_t)N * 128 * 2);
  float* q   = (float*)take((size_t)N * 128 * 4);
  float* kb  = (float*)take((size_t)N * 128 * 4);
  float* v   = (float*)take((size_t)N * 128 * 4);
  float* mx  = (float*)take((size_t)N * 128 * 4);
  float* z   = (float*)take((size_t)N * 128 * 4);
  float* num = (float*)take((size_t)N * 128 * 4);
  _Float16* yln  = (_Float16*)take((size_t)N * 128 * 2);
  _Float16* hmid = (_Float16*)take((size_t)N * 512 * 2);
  float* hb0 = (float*)take((size_t)N * 128 * 4);
  float* hb1 = (float*)take((size_t)N * 128 * 4);

  dim3 blk(256);
  k_f32_to_f16<<<(128 * 384 + 255) / 256, blk, 0, stream>>>(Wqkv, wqkv16, 128 * 384);
  k_f32_to_f16<<<(128 * 512 + 255) / 256, blk, 0, stream>>>(W1, w116, 128 * 512);
  k_f32_to_f16<<<(512 * 128 + 255) / 256, blk, 0, stream>>>(W2, w216, 512 * 128);

  const int NC = N * GDIM;
  for (int d = 0; d < 3; ++d) {
    const float* hin = (d == 0) ? h0 : ((d == 1) ? hb0 : hb1);
    float* hout = (d == 2) ? (float*)d_out : ((d == 0) ? hb0 : hb1);

    k_ln2<<<dim3((N + 7) / 8), blk, 0, stream>>>(hin, norm_w, norm_b, nin_w, nin_b, s16, N);
    k_wmma_gemm<<<dim3((N + 63) / 64, 384 / 128), blk, 0, stream>>>(
        s16, wqkv16, N, 128, 384, 0, nullptr, nullptr, q, kb, v, nullptr);
    k_init<<<(NC + 255) / 256, blk, 0, stream>>>(mx, z, num, NC);
    k_edge_max<<<(E + 7) / 8, blk, 0, stream>>>(q, kb, src, dst, mx, E);
    k_edge_acc<<<(E + 7) / 8, blk, 0, stream>>>(q, kb, v, src, dst, mx, z, num, E);
    k_finalize<<<dim3((N + 7) / 8), blk, 0, stream>>>(num, z, no_w, no_b, yln, N);
    k_wmma_gemm<<<dim3((N + 63) / 64, 512 / 128), blk, 0, stream>>>(
        yln, w116, N, 128, 512, 1, b1, nullptr, nullptr, nullptr, nullptr, hmid);
    k_wmma_gemm<<<dim3((N + 63) / 64, 128 / 128), blk, 0, stream>>>(
        hmid, w216, N, 512, 128, 2, b2, hin, hout, nullptr, nullptr, nullptr);
  }
}
